// DSTANet_59768764891350
// MI455X (gfx1250) — matrix operand order repaired and checked by reading, not compiled, not run on hardware
//
#include <hip/hip_runtime.h>
#include <hip/hip_bf16.h>
#include <cstdint>

typedef __attribute__((ext_vector_type(16))) __bf16 v16bf;
typedef __attribute__((ext_vector_type(8)))  float  v8f;
typedef __attribute__((ext_vector_type(4)))  float  v4f;

constexpr int NN = 64, CC = 128, CI = 32, SS = 3, TT = 64, VV = 25;
constexpr int PP = TT * VV;           // 1600 positions per (n)
constexpr float EPS = 1e-5f;

// ---------------------------------------------------------------------------
// CDNA5 async global->LDS copy (ASYNCcnt-tracked, VGPR-bypassing DMA path).
// lds_off: per-lane LDS byte offset; gaddr: per-lane 64-bit global address.
// ---------------------------------------------------------------------------
__device__ __forceinline__ void async_cp_b64(uint32_t lds_off, const float* gaddr) {
  asm volatile("global_load_async_to_lds_b64 %0, %1, off"
               :: "v"(lds_off), "v"(gaddr)
               : "memory");
}
__device__ __forceinline__ void wait_async_le2() {
  asm volatile("s_wait_asynccnt 0x2" ::: "memory");
}
__device__ __forceinline__ void wait_async_0() {
  asm volatile("s_wait_asynccnt 0x0" ::: "memory");
}

// ---------------------------------------------------------------------------
// y = x + pe  (pe broadcast over n)
// ---------------------------------------------------------------------------
__global__ void k_add_pe(const float* __restrict__ x, const float* __restrict__ pe,
                         float* __restrict__ y, int total, int ctv) {
  int i = blockIdx.x * blockDim.x + threadIdx.x;
  if (i < total) y[i] = x[i] + pe[i % ctv];
}

// ---------------------------------------------------------------------------
// Fused 1x1 conv GEMM via WMMA bf16 with async double-buffered LDS staging:
//   out[n,o,p] = epi( sum_c W[o,c] * in[n,c,p] + bias[o] )
//   epi: optional BN (g,b,m,v) then optional res-add + leaky_relu(0.1)
// Block = 128 threads = 4 waves; wave w computes rows [blockIdx.x*64+w*16,+16)
// over 16 columns p = blockIdx.y*16..+16, K stepped by 32 (bf16 WMMA).
// ---------------------------------------------------------------------------
template<int KDIM, bool BN, bool RES>
__global__ __launch_bounds__(128) void k_conv1x1_wmma(
    const float* __restrict__ in,   // (N, KDIM, PP) f32
    const float* __restrict__ Wt,   // (O, KDIM) f32 row-major
    const float* __restrict__ bias, // (O)
    const float* __restrict__ bg, const float* __restrict__ bb,
    const float* __restrict__ bm, const float* __restrict__ bv,
    const float* __restrict__ res,  // (N, O, PP) or null
    float* __restrict__ out,        // (N, O, PP)
    int O)
{
  constexpr int LSTR = 18; // padded LDS row stride (floats): 8B-aligned rows,
                           // banks (18h+p) vs (18h+p+16) mod 64 are disjoint
  __shared__ __attribute__((aligned(16))) float Bsf[2][32 * LSTR];

  const int tid   = threadIdx.x;
  const int lane  = tid & 31;
  const int wave  = tid >> 5;
  const int oBase = blockIdx.x * 64 + wave * 16;
  const int p0    = blockIdx.y * 16;
  const int n     = blockIdx.z;

  const float* inN = in + (size_t)n * KDIM * PP;
  const int m    = lane & 15;   // A row within tile (both lane halves)
  const int hi   = lane >> 4;   // lane group 0/1
  const int aoff = hi * 8;      // A K-offset per ISA 16-bit A 16x32 layout

  // staging assignment: thread -> (c row 0..31, p chunk of 4)
  const int sc  = tid >> 2;        // 0..31
  const int spc = (tid & 3) * 4;   // 0,4,8,12

  const float*   gsrc0   = inN + (size_t)sc * PP + p0 + spc;
  const uint32_t ldsrow  = (uint32_t)(uintptr_t)(&Bsf[0][sc * LSTR + spc]);
  const uint32_t bufstep = (uint32_t)(32 * LSTR * 4);

  // prologue: kick off tile 0 (2 x b64 per thread = full 32x16 f32 slab/block)
  async_cp_b64(ldsrow,     gsrc0);
  async_cp_b64(ldsrow + 8, gsrc0 + 2);

  v8f acc = {};
  constexpr int NSTEP = KDIM / 32;
  for (int step = 0; step < NSTEP; ++step) {
    const int cur = step & 1;
    if (step + 1 < NSTEP) {
      const float*   g = gsrc0 + (size_t)(step + 1) * 32 * PP;
      const uint32_t l = ldsrow + (uint32_t)((step + 1) & 1) * bufstep;
      async_cp_b64(l,     g);
      async_cp_b64(l + 8, g + 2);
      wait_async_le2();   // in-order completion -> tile[step]'s 2 ops done
    } else {
      wait_async_0();
    }
    __syncthreads();      // tile[step] visible to all 4 waves

    // ---- A fragment (weights): lanes 0-15 K=0..7,16..23; lanes 16-31 K=8..15,24..31
    const float* wrow = Wt + (size_t)(oBase + m) * KDIM + step * 32;
    const v4f w0 = *(const v4f*)(wrow + aoff);
    const v4f w1 = *(const v4f*)(wrow + aoff + 4);
    const v4f w2 = *(const v4f*)(wrow + 16 + aoff);
    const v4f w3 = *(const v4f*)(wrow + 16 + aoff + 4);
    v16bf a;
    #pragma unroll
    for (int j = 0; j < 4; ++j) {
      a[j]      = (__bf16)w0[j];
      a[4 + j]  = (__bf16)w1[j];
      a[8 + j]  = (__bf16)w2[j];
      a[12 + j] = (__bf16)w3[j];
    }
    // ---- B fragment: col = lane&15, K = hi*8 + 0..7, 16 + hi*8 + 0..7, f32->bf16 in regs
    const float* bbase = &Bsf[cur][hi * 8 * LSTR + (lane & 15)];
    v16bf b;
    #pragma unroll
    for (int h = 0; h < 16; h += 2) {
      b[h]     = (__bf16)bbase[(h + (h & 8)) * LSTR];
      b[h + 1] = (__bf16)bbase[(h + 1 + (h & 8)) * LSTR];
    }

    acc = __builtin_amdgcn_wmma_f32_16x16x32_bf16(false, a, false, b,
                                                  (short)0, acc, false, false);
    __syncthreads();      // all waves done reading buf[cur] before reuse
  }

  // ---- fused epilogue; D layout: VGPR r -> M = r + 8*hi, N = lane&15 ----
  const int pcol = p0 + (lane & 15);
  #pragma unroll
  for (int r = 0; r < 8; ++r) {
    const int o = oBase + r + 8 * hi;
    float val = acc[r] + bias[o];
    if constexpr (BN) {
      const float inv = bg[o] * rsqrtf(bv[o] + EPS);
      val = (val - bm[o]) * inv + bb[o];
    }
    const size_t oi = ((size_t)n * O + o) * PP + pcol;
    if constexpr (RES) {
      const float t = res[oi] + val;
      val = t > 0.f ? t : 0.1f * t;
    }
    out[oi] = val;
  }
}

// ---------------------------------------------------------------------------
// att_s[n,s,u,v] = tanh( sum_{c,t} q[n,s,c,t,u]*k[n,s,c,t,v] / 2048 )*alpha[s]
//                  + att0[s,u,v]
// ---------------------------------------------------------------------------
__global__ void k_att_spatial(const float* __restrict__ qk, const float* __restrict__ alpha,
                              const float* __restrict__ att0, float* __restrict__ att) {
  const int ns = blockIdx.x;
  const int n = ns / SS, s = ns % SS;
  const float* qb = qk + ((size_t)n * (2 * SS * CI) + s * CI) * PP;
  const float* kb = qk + ((size_t)n * (2 * SS * CI) + (SS + s) * CI) * PP;
  const float a = alpha[s];
  for (int uv = threadIdx.x; uv < VV * VV; uv += blockDim.x) {
    const int u = uv / VV, v = uv % VV;
    float acc = 0.f;
    for (int c = 0; c < CI; ++c) {
      const float* qr = qb + (size_t)c * PP + u;
      const float* kr = kb + (size_t)c * PP + v;
      #pragma unroll 4
      for (int t = 0; t < TT; ++t) acc += qr[t * VV] * kr[t * VV];
    }
    att[(size_t)ns * (VV * VV) + uv] =
        tanhf(acc * (1.f / (CI * TT))) * a + att0[s * VV * VV + uv];
  }
}

// se[n,s,c,t,v] = sum_u x[n,c,t,u] * att_s[n,s,u,v]
__global__ void k_apply_spatial(const float* __restrict__ x, const float* __restrict__ att,
                                float* __restrict__ se) {
  const size_t total = (size_t)NN * SS * CC * TT * VV;
  const size_t i = (size_t)blockIdx.x * blockDim.x + threadIdx.x;
  if (i >= total) return;
  const int v = (int)(i % VV);
  size_t r = i / VV;
  const int t = (int)(r % TT); r /= TT;
  const int c = (int)(r % CC); r /= CC;
  const int s = (int)(r % SS); const int n = (int)(r / SS);
  const float* xr = x + (((size_t)n * CC + c) * TT + t) * VV;
  const float* ar = att + (size_t)(n * SS + s) * (VV * VV) + v;
  float acc = 0.f;
  #pragma unroll
  for (int u = 0; u < VV; ++u) acc += xr[u] * ar[u * VV];
  se[i] = acc;
}

// att_t[n,s,t,q] = tanh( sum_{c,v} q[n,s,c,t,v]*k[n,s,c,q,v] / 800 )*alpha[s]
//                  + att0t[s,t,q]
__global__ void k_att_temporal(const float* __restrict__ qk, const float* __restrict__ alpha,
                               const float* __restrict__ att0, float* __restrict__ att) {
  const size_t total = (size_t)NN * SS * TT * TT;
  const size_t i = (size_t)blockIdx.x * blockDim.x + threadIdx.x;
  if (i >= total) return;
  const int q = (int)(i % TT);
  size_t r = i / TT;
  const int t = (int)(r % TT); r /= TT;
  const int s = (int)(r % SS); const int n = (int)(r / SS);
  const float* qb = qk + ((size_t)n * (2 * SS * CI) + s * CI) * PP + t * VV;
  const float* kb = qk + ((size_t)n * (2 * SS * CI) + (SS + s) * CI) * PP + q * VV;
  float acc = 0.f;
  for (int c = 0; c < CI; ++c) {
    const float* qr = qb + (size_t)c * PP;
    const float* kr = kb + (size_t)c * PP;
    #pragma unroll
    for (int v = 0; v < VV; ++v) acc += qr[v] * kr[v];
  }
  att[i] = tanhf(acc * (1.f / (CI * VV))) * alpha[s] + att0[(s * TT + t) * TT + q];
}

// se[n,s,c,q,v] = sum_t y[n,c,t,v] * att_t[n,s,t,q]
__global__ void k_apply_temporal(const float* __restrict__ y, const float* __restrict__ att,
                                 float* __restrict__ se) {
  const size_t total = (size_t)NN * SS * CC * TT * VV;
  const size_t i = (size_t)blockIdx.x * blockDim.x + threadIdx.x;
  if (i >= total) return;
  const int v = (int)(i % VV);
  size_t r = i / VV;
  const int q = (int)(r % TT); r /= TT;
  const int c = (int)(r % CC); r /= CC;
  const int s = (int)(r % SS); const int n = (int)(r / SS);
  const float* yr = y + ((size_t)n * CC + c) * TT * VV + v;
  const float* ar = att + (size_t)(n * SS + s) * TT * TT + q;
  float acc = 0.f;
  #pragma unroll 8
  for (int t = 0; t < TT; ++t) acc += yr[t * VV] * ar[t * TT];
  se[i] = acc;
}

// ---------------------------------------------------------------------------
extern "C" void kernel_launch(void* const* d_in, const int* in_sizes, int n_in,
                              void* d_out, int out_size, void* d_ws, size_t ws_size,
                              hipStream_t stream) {
  (void)in_sizes; (void)n_in; (void)out_size; (void)ws_size;
  const float* x       = (const float*)d_in[0];
  const float* pe_s    = (const float*)d_in[1];
  const float* pe_t    = (const float*)d_in[2];
  const float* w_in_s  = (const float*)d_in[3];
  const float* b_in_s  = (const float*)d_in[4];
  const float* alphas  = (const float*)d_in[5];
  const float* att0s   = (const float*)d_in[6];
  const float* w_out_s = (const float*)d_in[7];
  const float* b_out_s = (const float*)d_in[8];
  const float* w_ff_s  = (const float*)d_in[9];
  const float* b_ff_s  = (const float*)d_in[10];
  const float* w_in_t  = (const float*)d_in[11];
  const float* b_in_t  = (const float*)d_in[12];
  const float* alphat  = (const float*)d_in[13];
  const float* att0t   = (const float*)d_in[14];
  const float* w_out_t = (const float*)d_in[15];
  const float* b_out_t = (const float*)d_in[16];
  const float* w_ff_t  = (const float*)d_in[17];
  const float* b_ff_t  = (const float*)d_in[18];
  const float* g_out_s = (const float*)d_in[19];
  const float* be_out_s= (const float*)d_in[20];
  const float* m_out_s = (const float*)d_in[21];
  const float* v_out_s = (const float*)d_in[22];
  const float* g_ff_s  = (const float*)d_in[23];
  const float* be_ff_s = (const float*)d_in[24];
  const float* m_ff_s  = (const float*)d_in[25];
  const float* v_ff_s  = (const float*)d_in[26];
  const float* g_out_t = (const float*)d_in[27];
  const float* be_out_t= (const float*)d_in[28];
  const float* m_out_t = (const float*)d_in[29];
  const float* v_out_t = (const float*)d_in[30];
  const float* g_ff_t  = (const float*)d_in[31];
  const float* be_ff_t = (const float*)d_in[32];
  const float* m_ff_t  = (const float*)d_in[33];
  const float* v_ff_t  = (const float*)d_in[34];
  float* out = (float*)d_out;

  const size_t ACT = (size_t)NN * CC * TT * VV;            // 13,107,200
  const size_t QK  = (size_t)NN * 2 * SS * CI * TT * VV;   // 19,660,800
  const size_t SE  = (size_t)NN * SS * CC * TT * VV;       // 39,321,600

  float* bufY  = (float*)d_ws;         // PE-added activations
  float* bufQK = bufY  + ACT;          // q|k (N,192,P)
  float* bufSE = bufQK + QK;           // attention-expanded (N,384,P)
  float* bufY1 = bufSE + SE;           // post-out conv activations
  float* bufY2 = bufY1 + ACT;          // post-ff conv activations (spatial out)
  float* bufAS = bufY2 + ACT;          // att_s (N,S,V,V)
  float* bufAT = bufAS + (size_t)NN * SS * VV * VV; // att_t (N,S,T,T)

  const int CTV = CC * TT * VV;
  dim3 ew((unsigned)((ACT + 255) / 256));
  dim3 conv_in(3, PP / 16, NN);  // O = 192
  dim3 conv_o (2, PP / 16, NN);  // O = 128
  dim3 gse((unsigned)((SE + 255) / 256));
  dim3 gat((unsigned)(((size_t)NN * SS * TT * TT + 255) / 256));

  // ---------------- spatial branch ----------------
  k_add_pe<<<ew, 256, 0, stream>>>(x, pe_s, bufY, (int)ACT, CTV);
  k_conv1x1_wmma<128, false, false><<<conv_in, 128, 0, stream>>>(
      bufY, w_in_s, b_in_s, nullptr, nullptr, nullptr, nullptr, nullptr, bufQK, 192);
  k_att_spatial<<<NN * SS, 128, 0, stream>>>(bufQK, alphas, att0s, bufAS);
  k_apply_spatial<<<gse, 256, 0, stream>>>(x, bufAS, bufSE);
  k_conv1x1_wmma<384, true, true><<<conv_o, 128, 0, stream>>>(
      bufSE, w_out_s, b_out_s, g_out_s, be_out_s, m_out_s, v_out_s, x, bufY1, 128);
  k_conv1x1_wmma<128, true, true><<<conv_o, 128, 0, stream>>>(
      bufY1, w_ff_s, b_ff_s, g_ff_s, be_ff_s, m_ff_s, v_ff_s, x, bufY2, 128);

  // ---------------- temporal branch ----------------
  k_add_pe<<<ew, 256, 0, stream>>>(bufY2, pe_t, bufY, (int)ACT, CTV);
  k_conv1x1_wmma<128, false, false><<<conv_in, 128, 0, stream>>>(
      bufY, w_in_t, b_in_t, nullptr, nullptr, nullptr, nullptr, nullptr, bufQK, 192);
  k_att_temporal<<<gat, 256, 0, stream>>>(bufQK, alphat, att0t, bufAT);
  k_apply_temporal<<<gse, 256, 0, stream>>>(bufY2, bufAT, bufSE);
  k_conv1x1_wmma<384, true, true><<<conv_o, 128, 0, stream>>>(
      bufSE, w_out_t, b_out_t, g_out_t, be_out_t, m_out_t, v_out_t, bufY2, bufY1, 128);
  k_conv1x1_wmma<128, true, true><<<conv_o, 128, 0, stream>>>(
      bufY1, w_ff_t, b_ff_t, g_ff_t, be_ff_t, m_ff_t, v_ff_t, bufY2, out, 128);
}
